// SimpleMambaModel_44633300140706
// MI455X (gfx1250) — matrix operand order repaired and checked by reading, not compiled, not run on hardware
//
#include <hip/hip_runtime.h>
#include <hip/hip_bf16.h>

typedef __bf16 bf16_t;
typedef __attribute__((ext_vector_type(16))) __bf16 v16bf;
typedef __attribute__((ext_vector_type(8)))  __bf16 v8bf;
typedef __attribute__((ext_vector_type(8)))  float   v8f;

#define BB      2
#define LL      1024
#define DM      768
#define DIN     1536
#define NST     16
#define DTR     48
#define VOCABSZ 32000
#define ROWS    (BB * LL)        // 2048
#define XPN     (DTR + 2 * NST)  // 80
#define TPAD    40               // LDS row stride (elements) for a 32-wide K tile

// ---------------------------------------------------------------------------
// LDS helpers: 32-bit LDS byte offset for async-DMA, async copy, waits
// ---------------------------------------------------------------------------
__device__ __forceinline__ unsigned lds_off(const void* p) {
  return (unsigned)(unsigned long long)
      ((const __attribute__((address_space(3))) char*)p);
}

// One wave-instruction: every lane copies 16B global -> LDS (ASYNCcnt path)
__device__ __forceinline__ void async_cp16(const bf16_t* g, const bf16_t* s) {
  asm volatile("global_load_async_to_lds_b128 %0, %1, off"
               :: "v"(lds_off(s)), "v"((unsigned long long)g)
               : "memory");
}

__device__ __forceinline__ void wait_async_le4() {
  asm volatile("s_wait_asynccnt 0x4" ::: "memory");
}
__device__ __forceinline__ void wait_async_0() {
  asm volatile("s_wait_asynccnt 0x0" ::: "memory");
}

// ---------------------------------------------------------------------------
// WMMA fragment loaders from LDS tiles (wave32; layouts per CDNA5 ISA 7.12.2)
// A 16x32 bf16: lane r holds row r, K = {8h..8h+8} and {16+8h..16+8h+8}
// B 32x16 bf16: lane n holds col n, K = {8h..8h+8} and {16+8h..16+8h+8}
// ---------------------------------------------------------------------------
__device__ __forceinline__ v16bf frag_a(const bf16_t* t, int row, int half) {
  const bf16_t* p = t + row * TPAD + 8 * half;
  v8bf lo = *(const v8bf*)p;
  v8bf hi = *(const v8bf*)(p + 16);
  v16bf r;
#pragma unroll
  for (int i = 0; i < 8; ++i) { r[i] = lo[i]; r[i + 8] = hi[i]; }
  return r;
}

__device__ __forceinline__ v16bf frag_b(const bf16_t* t, int row, int half) {
  const bf16_t* p = t + row * TPAD + 8 * half;
  v8bf lo = *(const v8bf*)p;
  v8bf hi = *(const v8bf*)(p + 16);
  v16bf r;
#pragma unroll
  for (int i = 0; i < 8; ++i) { r[i] = lo[i]; r[i + 8] = hi[i]; }
  return r;
}

// ---------------------------------------------------------------------------
// Stage one 128x32 A tile + 128x32 B tile into LDS via async DMA.
// 512 16B-chunks per tile; 256 threads x 2 -> 4 async instrs per thread.
// OOB rows are clamped (their outputs are never stored).
// ---------------------------------------------------------------------------
__device__ __forceinline__ void stage(const bf16_t* __restrict__ A,
                                      const bf16_t* __restrict__ Bw,
                                      int M, int N, int K,
                                      int mTile, int nTile, int k0,
                                      bf16_t* sA, bf16_t* sB, int tid) {
#pragma unroll
  for (int j = 0; j < 2; ++j) {
    int c   = j * 256 + tid;       // 0..511
    int row = c >> 2;              // 0..127
    int sub = c & 3;               // 16B chunk within 64B row-slice
    int ga = mTile + row; ga = (ga > M - 1) ? (M - 1) : ga;
    int gb = nTile + row; gb = (gb > N - 1) ? (N - 1) : gb;
    async_cp16(A  + (long)ga * K + k0 + sub * 8, sA + row * TPAD + sub * 8);
    async_cp16(Bw + (long)gb * K + k0 + sub * 8, sB + row * TPAD + sub * 8);
  }
}

// ---------------------------------------------------------------------------
// GEMM: C[M,N] = A[M,K(bf16)] * W[N,K(bf16)]^T (+ Cadd), fp32 accum.
// Block = 256 threads (8 waves). Tile 128x128. Wave grid 4(M) x 2(N);
// each wave computes 32x64 via 8 v_wmma_f32_16x16x32_bf16 per K-step of 32.
// Double-buffered async global->LDS staging. K must be a multiple of 32.
// ---------------------------------------------------------------------------
__global__ __launch_bounds__(256) void k_gemm_bf16(
    const bf16_t* __restrict__ A, const bf16_t* __restrict__ Bw,
    float* __restrict__ C, const float* __restrict__ Cadd,
    int M, int N, int K) {
  __shared__ __align__(16) bf16_t sA[2][128 * TPAD];
  __shared__ __align__(16) bf16_t sB[2][128 * TPAD];

  const int tid  = threadIdx.x;
  const int lane = tid & 31;
  const int wave = tid >> 5;
  const int wm   = wave & 3;        // 32-row M slice within tile
  const int wn   = wave >> 2;       // 64-col N slice within tile
  const int mTile = blockIdx.y * 128;
  const int nTile = blockIdx.x * 128;
  const int r    = lane & 15;
  const int half = lane >> 4;

  v8f acc[2][4] = {};

  stage(A, Bw, M, N, K, mTile, nTile, 0, sA[0], sB[0], tid);

  const int T = K >> 5;
  for (int kt = 0; kt < T; ++kt) {
    const int cur = kt & 1;
    if (kt + 1 < T) {
      stage(A, Bw, M, N, K, mTile, nTile, (kt + 1) << 5,
            sA[cur ^ 1], sB[cur ^ 1], tid);
      wait_async_le4();            // current stage complete; next in flight
    } else {
      wait_async_0();
    }
    __syncthreads();               // tile visible to all waves

    const bf16_t* tA = sA[cur];
    const bf16_t* tB = sB[cur];
    v16bf a0 = frag_a(tA, 32 * wm + r,      half);
    v16bf a1 = frag_a(tA, 32 * wm + 16 + r, half);
    v16bf b0 = frag_b(tB, 64 * wn + r,      half);
    v16bf b1 = frag_b(tB, 64 * wn + 16 + r, half);
    v16bf b2 = frag_b(tB, 64 * wn + 32 + r, half);
    v16bf b3 = frag_b(tB, 64 * wn + 48 + r, half);

    acc[0][0] = __builtin_amdgcn_wmma_f32_16x16x32_bf16(false, a0, false, b0, (short)0, acc[0][0], false, false);
    acc[0][1] = __builtin_amdgcn_wmma_f32_16x16x32_bf16(false, a0, false, b1, (short)0, acc[0][1], false, false);
    acc[0][2] = __builtin_amdgcn_wmma_f32_16x16x32_bf16(false, a0, false, b2, (short)0, acc[0][2], false, false);
    acc[0][3] = __builtin_amdgcn_wmma_f32_16x16x32_bf16(false, a0, false, b3, (short)0, acc[0][3], false, false);
    acc[1][0] = __builtin_amdgcn_wmma_f32_16x16x32_bf16(false, a1, false, b0, (short)0, acc[1][0], false, false);
    acc[1][1] = __builtin_amdgcn_wmma_f32_16x16x32_bf16(false, a1, false, b1, (short)0, acc[1][1], false, false);
    acc[1][2] = __builtin_amdgcn_wmma_f32_16x16x32_bf16(false, a1, false, b2, (short)0, acc[1][2], false, false);
    acc[1][3] = __builtin_amdgcn_wmma_f32_16x16x32_bf16(false, a1, false, b3, (short)0, acc[1][3], false, false);

    __syncthreads();               // all reads done before buffer reuse
  }

  // C/D layout: lane -> N = lane&15; VGPR v -> M = v + 8*(lane>>4)
  const int lane15 = lane & 15;
  const int mh     = (lane >> 4) * 8;
#pragma unroll
  for (int i = 0; i < 2; ++i) {
#pragma unroll
    for (int j = 0; j < 4; ++j) {
      int nC = nTile + 64 * wn + 16 * j + lane15;
      int mB = mTile + 32 * wm + 16 * i + mh;
      if (nC < N) {
#pragma unroll
        for (int v = 0; v < 8; ++v) {
          int m = mB + v;
          if (m < M) {
            long idx = (long)m * N + nC;
            C[idx] = Cadd ? (acc[i][j][v] + Cadd[idx]) : acc[i][j][v];
          }
        }
      }
    }
  }
}

// ---------------------------------------------------------------------------
// fp32 -> bf16 with optional K padding (zeros). src row stride = sstride.
// ---------------------------------------------------------------------------
__global__ void k_cvt(const float* __restrict__ src, bf16_t* __restrict__ dst,
                      int rows, int scols, int sstride, int dcols) {
  long i = (long)blockIdx.x * blockDim.x + threadIdx.x;
  long total = (long)rows * dcols;
  if (i >= total) return;
  int rr = (int)(i / dcols);
  int c  = (int)(i - (long)rr * dcols);
  float v = (c < scols) ? src[(long)rr * sstride + c] : 0.f;
  dst[i] = (bf16_t)v;
}

// ---------------------------------------------------------------------------
// Embedding gather: h[row,:] = embedding[x[row],:]
// ---------------------------------------------------------------------------
__global__ __launch_bounds__(256) void k_embed(const int* __restrict__ x,
                                               const float* __restrict__ emb,
                                               float* __restrict__ h) {
  int row = blockIdx.x;
  int tok = x[row];
  for (int c = threadIdx.x; c < DM; c += 256)
    h[(long)row * DM + c] = emb[(long)tok * DM + c];
}

// ---------------------------------------------------------------------------
// Causal depthwise conv (K=4) + bias + SiLU. x_in = xz[:, 0:DIN].
// ---------------------------------------------------------------------------
__global__ __launch_bounds__(256) void k_conv_silu(
    const float* __restrict__ xz, const float* __restrict__ cw,
    const float* __restrict__ cb, float* __restrict__ xc,
    bf16_t* __restrict__ xcb) {
  long i = (long)blockIdx.x * blockDim.x + threadIdx.x;
  if (i >= (long)ROWS * DIN) return;
  int d   = (int)(i % DIN);
  long bl = i / DIN;           // b*L + l
  int l   = (int)(bl % LL);
  float acc = cb[d];
#pragma unroll
  for (int k = 0; k < 4; ++k) {
    int ll = l - 3 + k;
    if (ll >= 0) acc += cw[d * 4 + k] * xz[(bl - 3 + k) * (2 * DIN) + d];
  }
  float s = acc / (1.f + __expf(-acc));
  xc[i]  = s;
  xcb[i] = (bf16_t)s;
}

// ---------------------------------------------------------------------------
// dt = softplus(dt_raw + dt_bias)   (in place)
// ---------------------------------------------------------------------------
__global__ __launch_bounds__(256) void k_softplus(float* __restrict__ dt,
                                                  const float* __restrict__ dtb) {
  long i = (long)blockIdx.x * blockDim.x + threadIdx.x;
  if (i >= (long)ROWS * DIN) return;
  int d = (int)(i % DIN);
  float t = dt[i] + dtb[d];
  dt[i] = (t > 20.f) ? t : log1pf(__expf(t));
}

// ---------------------------------------------------------------------------
// Selective scan. One thread per (b, d, n); wave32 = 2 channels x 16 states.
// h_n(t) = exp(dt*A_n)*h_n(t-1) + dt*B_n*u ; y = sum_n h_n*C_n (shfl reduce).
// ---------------------------------------------------------------------------
__global__ __launch_bounds__(256) void k_scan(
    const float* __restrict__ dt, const float* __restrict__ u,
    const float* __restrict__ xdbl, const float* __restrict__ Alog,
    float* __restrict__ ys) {
  const int lane = threadIdx.x & 31;
  const int wave = threadIdx.x >> 5;
  const int n    = lane & 15;
  const int c    = blockIdx.x * 16 + wave * 2 + (lane >> 4);  // 0..BB*DIN-1
  const int b    = c / DIN;
  const int d    = c - b * DIN;

  const float a = -__expf(Alog[d * NST + n]);
  float h = 0.f;
  for (int l = 0; l < LL; ++l) {
    const long row = (long)b * LL + l;
    const float dtv = dt[row * DIN + d];
    const float uv  = u [row * DIN + d];
    const float bn  = xdbl[row * XPN + DTR + n];
    const float cn  = xdbl[row * XPN + DTR + NST + n];
    h = __expf(dtv * a) * h + (dtv * bn) * uv;
    float p = h * cn;
    p += __shfl_xor(p, 1);
    p += __shfl_xor(p, 2);
    p += __shfl_xor(p, 4);
    p += __shfl_xor(p, 8);
    if (n == 0) ys[row * DIN + d] = p;
  }
}

// ---------------------------------------------------------------------------
// y = (scan_y + u*D) * silu(z);  z = xz[:, DIN:2*DIN]; output bf16
// ---------------------------------------------------------------------------
__global__ __launch_bounds__(256) void k_gate(
    const float* __restrict__ ys, const float* __restrict__ xc,
    const float* __restrict__ xz, const float* __restrict__ Dp,
    bf16_t* __restrict__ yb) {
  long i = (long)blockIdx.x * blockDim.x + threadIdx.x;
  if (i >= (long)ROWS * DIN) return;
  int d    = (int)(i % DIN);
  long row = i / DIN;
  float z   = xz[row * (2 * DIN) + DIN + d];
  float sil = z / (1.f + __expf(-z));
  yb[i] = (bf16_t)((ys[i] + xc[i] * Dp[d]) * sil);
}

// ---------------------------------------------------------------------------
// LayerNorm over 768; write bf16 (input to tied lm_head GEMM).
// ---------------------------------------------------------------------------
__global__ __launch_bounds__(256) void k_layernorm(
    const float* __restrict__ h, const float* __restrict__ w,
    const float* __restrict__ bb, bf16_t* __restrict__ outb) {
  __shared__ float red[256];
  const int row = blockIdx.x;
  const int t   = threadIdx.x;
  const float* hr = h + (long)row * DM;

  float s = 0.f;
  for (int c = t; c < DM; c += 256) s += hr[c];
  red[t] = s; __syncthreads();
  for (int o = 128; o > 0; o >>= 1) { if (t < o) red[t] += red[t + o]; __syncthreads(); }
  float mu = red[0] / DM;
  __syncthreads();

  float q = 0.f;
  for (int c = t; c < DM; c += 256) { float dd = hr[c] - mu; q += dd * dd; }
  red[t] = q; __syncthreads();
  for (int o = 128; o > 0; o >>= 1) { if (t < o) red[t] += red[t + o]; __syncthreads(); }
  float rstd = rsqrtf(red[0] / DM + 1e-5f);

  for (int c = t; c < DM; c += 256)
    outb[(long)row * DM + c] = (bf16_t)((hr[c] - mu) * rstd * w[c] + bb[c]);
}

// ===========================================================================
extern "C" void kernel_launch(void* const* d_in, const int* in_sizes, int n_in,
                              void* d_out, int out_size, void* d_ws, size_t ws_size,
                              hipStream_t stream) {
  (void)in_sizes; (void)n_in; (void)out_size; (void)ws_size;
  const int*   x      = (const int*)  d_in[0];
  const float* emb    = (const float*)d_in[1];
  const float* w_in   = (const float*)d_in[2];
  const float* conv_w = (const float*)d_in[3];
  const float* conv_b = (const float*)d_in[4];
  const float* w_xp   = (const float*)d_in[5];
  const float* w_dt   = (const float*)d_in[6];
  const float* dt_b   = (const float*)d_in[7];
  const float* A_log  = (const float*)d_in[8];
  const float* Dp     = (const float*)d_in[9];
  const float* w_out  = (const float*)d_in[10];
  const float* ln_w   = (const float*)d_in[11];
  const float* ln_b   = (const float*)d_in[12];
  float* out = (float*)d_out;

  // ---- carve workspace (256B aligned slabs) ----
  char* p = (char*)d_ws;
  auto take = [&](size_t elems, size_t esz) -> void* {
    void* r = (void*)p;
    p += (elems * esz + 255) & ~(size_t)255;
    return r;
  };
  bf16_t* emb_b = (bf16_t*)take((size_t)VOCABSZ * DM, 2);
  bf16_t *win_b[2], *wxp_b[2], *wdt_b[2], *wout_b[2];
  for (int l = 0; l < 2; ++l) {
    win_b[l]  = (bf16_t*)take((size_t)2 * DIN * DM, 2);
    wxp_b[l]  = (bf16_t*)take((size_t)XPN * DIN, 2);
    wdt_b[l]  = (bf16_t*)take((size_t)DIN * 64, 2);   // K padded 48 -> 64
    wout_b[l] = (bf16_t*)take((size_t)DM * DIN, 2);
  }
  float*  h    = (float*) take((size_t)ROWS * DM, 4);
  bf16_t* h_b  = (bf16_t*)take((size_t)ROWS * DM, 2);
  float*  xz   = (float*) take((size_t)ROWS * 2 * DIN, 4);
  float*  xc   = (float*) take((size_t)ROWS * DIN, 4);
  bf16_t* xc_b = (bf16_t*)take((size_t)ROWS * DIN, 2);
  float*  xdbl = (float*) take((size_t)ROWS * XPN, 4);
  bf16_t* dts_b= (bf16_t*)take((size_t)ROWS * 64, 2); // dt slice, K padded
  float*  dtf  = (float*) take((size_t)ROWS * DIN, 4);
  float*  ys   = (float*) take((size_t)ROWS * DIN, 4);
  bf16_t* y_b  = (bf16_t*)take((size_t)ROWS * DIN, 2);
  bf16_t* hn_b = (bf16_t*)take((size_t)ROWS * DM, 2);

  auto cvt = [&](const float* src, bf16_t* dst, int rows, int scols, int sstride,
                 int dcols) {
    long total = (long)rows * dcols;
    k_cvt<<<(unsigned)((total + 255) / 256), 256, 0, stream>>>(src, dst, rows,
                                                               scols, sstride, dcols);
  };
  auto gemm = [&](const bf16_t* Aa, const bf16_t* Bb, float* Cc, const float* Ca,
                  int M, int N, int K) {
    dim3 g((unsigned)((N + 127) / 128), (unsigned)((M + 127) / 128));
    k_gemm_bf16<<<g, dim3(256), 0, stream>>>(Aa, Bb, Cc, Ca, M, N, K);
  };

  // ---- weight conversions (deterministic, every call) ----
  cvt(emb, emb_b, VOCABSZ, DM, DM, DM);
  for (int l = 0; l < 2; ++l) {
    cvt(w_in  + (size_t)l * 2 * DIN * DM, win_b[l],  2 * DIN, DM,  DM,  DM);
    cvt(w_xp  + (size_t)l * XPN * DIN,    wxp_b[l],  XPN,     DIN, DIN, DIN);
    cvt(w_dt  + (size_t)l * DIN * DTR,    wdt_b[l],  DIN,     DTR, DTR, 64);
    cvt(w_out + (size_t)l * DM * DIN,     wout_b[l], DM,      DIN, DIN, DIN);
  }

  // ---- forward ----
  k_embed<<<ROWS, 256, 0, stream>>>(x, emb, h);

  const long ed = (long)ROWS * DIN;
  for (int l = 0; l < 2; ++l) {
    cvt(h, h_b, ROWS, DM, DM, DM);
    gemm(h_b, win_b[l], xz, nullptr, ROWS, 2 * DIN, DM);                 // in_proj
    k_conv_silu<<<(unsigned)((ed + 255) / 256), 256, 0, stream>>>(
        xz, conv_w + (size_t)l * DIN * 4, conv_b + (size_t)l * DIN, xc, xc_b);
    gemm(xc_b, wxp_b[l], xdbl, nullptr, ROWS, XPN, DIN);                 // x_proj
    cvt(xdbl, dts_b, ROWS, DTR, XPN, 64);                                // dt slice
    gemm(dts_b, wdt_b[l], dtf, nullptr, ROWS, DIN, 64);                  // dt_proj
    k_softplus<<<(unsigned)((ed + 255) / 256), 256, 0, stream>>>(
        dtf, dt_b + (size_t)l * DIN);
    k_scan<<<(BB * DIN) / 16, 256, 0, stream>>>(
        dtf, xc, xdbl, A_log + (size_t)l * DIN * NST, ys);
    k_gate<<<(unsigned)((ed + 255) / 256), 256, 0, stream>>>(
        ys, xc, xz, Dp + (size_t)l * DIN, y_b);
    gemm(y_b, wout_b[l], h, h, ROWS, DM, DIN);                           // out_proj + residual
  }

  k_layernorm<<<ROWS, 256, 0, stream>>>(h, ln_w, ln_b, hn_b);
  gemm(hn_b, emb_b, out, nullptr, ROWS, VOCABSZ, DM);                    // tied lm_head
}
